// KDGQA_20315195310698
// MI455X (gfx1250) — matrix operand order repaired and checked by reading, not compiled, not run on hardware
//
#include <hip/hip_runtime.h>
#include <hip/hip_bf16.h>

#define DIMC 1024
#define NH   16
#define NKV  8
#define HDIM 64
#define BBATCH 8
#define PSEQ 1024
#define QKVN 2048
#define MROWS (BBATCH*PSEQ)   // 8192

typedef __attribute__((ext_vector_type(16))) __bf16 v16bf;
typedef __attribute__((ext_vector_type(8)))  float  v8f;

union Frag { v16bf v; unsigned int u[8]; };

__device__ inline unsigned short f2bf(float f) {
    unsigned int u = __float_as_uint(f);
    u += 0x7FFFu + ((u >> 16) & 1u);          // round-to-nearest-even
    return (unsigned short)(u >> 16);
}
__device__ inline float bf2f(unsigned short h) {
    return __uint_as_float(((unsigned int)h) << 16);
}

// CDNA5 async global->LDS copy, 16B per lane, tracked by ASYNCcnt.
// lds_addr: low 32 bits of the flat shared-memory address (= LDS offset).
__device__ inline void async_copy_b128(const unsigned short* lds_ptr,
                                       const unsigned short* gptr) {
    unsigned int lds_addr = (unsigned int)(size_t)lds_ptr;
    asm volatile("global_load_async_to_lds_b128 %0, %1, off"
                 :: "v"(lds_addr), "v"(gptr) : "memory");
}
__device__ inline void wait_async0() {
    asm volatile("s_wait_asynccnt 0" ::: "memory");
}

// ---------------------------------------------------------------- converts
__global__ void kdgqa_cvt_bf16(const float* __restrict__ src,
                               unsigned short* __restrict__ dst, int n) {
    for (int i = blockIdx.x * blockDim.x + threadIdx.x; i < n;
         i += gridDim.x * blockDim.x)
        dst[i] = f2bf(src[i]);
}

// ---------------------------------------------------------------- generic GEMM
// C[M,N] = A[M,K] * B[N,K]^T  (A,B bf16 row-major, f32 accumulate)
// Block tile 128x128, 8 waves, each wave 4 M-tiles x 2 N-tiles (8 WMMA/K-step).
// Async double-buffered LDS staging.
__global__ __launch_bounds__(256)
void kdgqa_gemm_wmma(const unsigned short* __restrict__ A,
                     const unsigned short* __restrict__ Bw,
                     unsigned short* __restrict__ Cbf,
                     float* __restrict__ Cf,
                     const float* __restrict__ bias,
                     int M, int N, int K, int ncut, float scale) {
    __shared__ unsigned short Asl[2][128 * 32];
    __shared__ unsigned short Bsl[2][128 * 32];
    const int t = threadIdx.x;
    const int wave = t >> 5, lane = t & 31;
    const int hi = lane >> 4, ln = lane & 15;
    const int n0 = blockIdx.x * 128, m0 = blockIdx.y * 128;
    const int wn = (wave & 3) * 32;       // 2 N-tiles: wn, wn+16
    const int wm = (wave >> 2) * 64;      // 4 M-tiles: wm, wm+16, wm+32, wm+48

    // loader mapping: 128 rows x 32 ushort = 512 16B-segments; 2 per thread
    const int r0 = t >> 1;                // segments t and t+256 -> rows t>>1, same pair cols
    const int c0 = (t & 1) * 16;          // hmm: use simple 2-seg split below
    (void)r0; (void)c0;

    v8f acc[4][2] = {};

    auto issue = [&](int buf, int k0) {
#pragma unroll
        for (int s = 0; s < 2; ++s) {
            int seg = t + s * 256;        // 0..511
            int row = seg >> 2;           // 0..127
            int col = (seg & 3) * 8;      // ushort col
            async_copy_b128(&Asl[buf][row * 32 + col],
                            &A[(size_t)(m0 + row) * K + k0 + col]);
            async_copy_b128(&Bsl[buf][row * 32 + col],
                            &Bw[(size_t)(n0 + row) * K + k0 + col]);
        }
    };

    issue(0, 0);
    int buf = 0;
    for (int k0 = 0; k0 < K; k0 += 32) {
        wait_async0();
        __syncthreads();
        if (k0 + 32 < K) issue(buf ^ 1, k0 + 32);

        Frag af[4], bf_[2];
#pragma unroll
        for (int i = 0; i < 4; ++i)
#pragma unroll
            for (int j = 0; j < 8; ++j) {
                int ka = 2 * j + ((j >= 4) ? 8 : 0) + (hi ? 8 : 0);
                af[i].u[j] = *(const unsigned int*)&Asl[buf][(wm + 16 * i + ln) * 32 + ka];
            }
#pragma unroll
        for (int i = 0; i < 2; ++i)
#pragma unroll
            for (int j = 0; j < 8; ++j) {
                int kb = 2 * j + ((j >= 4) ? 8 : 0) + (hi ? 8 : 0);
                bf_[i].u[j] = *(const unsigned int*)&Bsl[buf][(wn + 16 * i + ln) * 32 + kb];
            }
#pragma unroll
        for (int i = 0; i < 4; ++i)
#pragma unroll
            for (int j = 0; j < 2; ++j)
                acc[i][j] = __builtin_amdgcn_wmma_f32_16x16x32_bf16(
                    false, af[i].v, false, bf_[j].v, (short)0, acc[i][j], false, false);
        buf ^= 1;
    }

#pragma unroll
    for (int i = 0; i < 4; ++i)
#pragma unroll
        for (int j = 0; j < 2; ++j)
#pragma unroll
            for (int e = 0; e < 8; ++e) {
                int col = n0 + wn + 16 * j + ln;
                int row = m0 + wm + 16 * i + e + hi * 8;
                float v = acc[i][j][e];
                if (Cf) {
                    Cf[(size_t)row * N + col] = v + (bias ? bias[col] : 0.0f);
                } else {
                    float s = (col < ncut) ? scale : 1.0f;
                    Cbf[(size_t)row * N + col] = f2bf(v * s);
                }
            }
}

// ---------------------------------------------------------------- k-norm partials
__global__ __launch_bounds__(256)
void kdgqa_knorm(const unsigned short* __restrict__ qkv, float* __restrict__ part) {
    int b = blockIdx.x >> 3;
    int g = blockIdx.x & 7;
    float s = 0.f;
    for (int idx = threadIdx.x; idx < PSEQ * HDIM; idx += 256) {
        int p = idx >> 6, d = idx & 63;
        float v = bf2f(qkv[(size_t)(b * PSEQ + p) * QKVN + DIMC + g * HDIM + d]);
        s += v * v;
    }
    __shared__ float red[256];
    red[threadIdx.x] = s;
    __syncthreads();
    for (int off = 128; off > 0; off >>= 1) {
        if (threadIdx.x < off) red[threadIdx.x] += red[threadIdx.x + off];
        __syncthreads();
    }
    if (threadIdx.x == 0) part[blockIdx.x] = red[0];
}

// ---------------------------------------------------------------- allocation logic
__global__ void kdgqa_alloc(const float* __restrict__ part, int* __restrict__ kvidx) {
    if (threadIdx.x != 0) return;
    float kn[NKV];
    for (int g = 0; g < NKV; ++g) {
        float s = 0.f;
        for (int b = 0; b < BBATCH; ++b) s += sqrtf(part[b * NKV + g]);
        kn[g] = s;
    }
    float mn = kn[0], mx = kn[0];
    for (int g = 1; g < NKV; ++g) { mn = fminf(mn, kn[g]); mx = fmaxf(mx, kn[g]); }
    float den = mx - mn; if (den == 0.f) den = 1.f;
    float sum = 0.f;
    for (int g = 0; g < NKV; ++g) { kn[g] = (kn[g] - mn) / den; sum += kn[g]; }
    if (sum == 0.f) sum = 1.f;
    int alloc[NKV]; int tot = 0;
    for (int g = 0; g < NKV; ++g) { alloc[g] = (int)rintf(kn[g] / sum * NH); tot += alloc[g]; }
    while (tot > NH) { int a = 0; for (int g = 1; g < NKV; ++g) if (alloc[g] > alloc[a]) a = g; alloc[a]--; tot--; }
    while (tot < NH) { int a = 0; for (int g = 1; g < NKV; ++g) if (alloc[g] < alloc[a]) a = g; alloc[a]++; tot++; }
    int cum = 0, h = 0;
    for (int g = 0; g < NKV; ++g) {
        cum += alloc[g];
        while (h < cum && h < NH) kvidx[h++] = g;
    }
    while (h < NH) kvidx[h++] = NKV - 1;
}

// ---------------------------------------------------------------- flash attention
__global__ __launch_bounds__(256)
void kdgqa_attn_wmma(const unsigned short* __restrict__ qkv,
                     const int* __restrict__ kvidx,
                     unsigned short* __restrict__ attn_out) {
    __shared__ unsigned short Klds[2][32 * 64];
    __shared__ unsigned short Vlds[2][32 * 64];
    __shared__ unsigned short Plds[8 * 16 * 32];

    const int t = threadIdx.x;
    const int wave = t >> 5, lane = t & 31;
    const int hi = lane >> 4, ln = lane & 15;

    const int bid = blockIdx.x;       // 8 * 16 * 8 = 1024 blocks
    const int qc = bid & 7;
    const int h  = (bid >> 3) & 15;
    const int b  = bid >> 7;
    const int g  = kvidx[h];

    const int q0 = qc * 128 + wave * 16;

    auto issue_kv = [&](int buf, int kt) {
        int row = t >> 3;             // 0..31
        int col = (t & 7) * 8;        // ushort col 0..56
        size_t gr = ((size_t)(b * PSEQ + kt + row)) * QKVN + DIMC + g * HDIM + col;
        async_copy_b128(&Klds[buf][row * 64 + col], &qkv[gr]);
        async_copy_b128(&Vlds[buf][row * 64 + col], &qkv[gr + 512]);
    };

    Frag qa0, qa1;
    {
        size_t rb = ((size_t)(b * PSEQ + q0 + ln)) * QKVN + h * HDIM;
#pragma unroll
        for (int j = 0; j < 8; ++j) {
            int ka = 2 * j + ((j >= 4) ? 8 : 0) + (hi ? 8 : 0);
            qa0.u[j] = *(const unsigned int*)&qkv[rb + ka];
            qa1.u[j] = *(const unsigned int*)&qkv[rb + 32 + ka];
        }
    }

    float mrow[8], lrow[8];
#pragma unroll
    for (int i = 0; i < 8; ++i) { mrow[i] = -3.0e38f; lrow[i] = 0.f; }
    v8f o0 = {}, o1 = {}, o2 = {}, o3 = {};

    issue_kv(0, 0);
    int buf = 0;
    for (int kt = 0; kt < PSEQ; kt += 32) {
        wait_async0();
        __syncthreads();
        if (kt + 32 < PSEQ) issue_kv(buf ^ 1, kt + 32);

        v8f s0 = {}, s1 = {};
#pragma unroll
        for (int half = 0; half < 2; ++half) {
            Frag kb0, kb1;
#pragma unroll
            for (int j = 0; j < 8; ++j) {
                int kk = 2 * j + ((j >= 4) ? 8 : 0) + (hi ? 8 : 0);
                kb0.u[j] = *(const unsigned int*)&Klds[buf][(half * 16 + ln) * 64 + kk];
                kb1.u[j] = *(const unsigned int*)&Klds[buf][(half * 16 + ln) * 64 + 32 + kk];
            }
            if (half == 0) {
                s0 = __builtin_amdgcn_wmma_f32_16x16x32_bf16(false, qa0.v, false, kb0.v,
                                                             (short)0, s0, false, false);
                s0 = __builtin_amdgcn_wmma_f32_16x16x32_bf16(false, qa1.v, false, kb1.v,
                                                             (short)0, s0, false, false);
            } else {
                s1 = __builtin_amdgcn_wmma_f32_16x16x32_bf16(false, qa0.v, false, kb0.v,
                                                             (short)0, s1, false, false);
                s1 = __builtin_amdgcn_wmma_f32_16x16x32_bf16(false, qa1.v, false, kb1.v,
                                                             (short)0, s1, false, false);
            }
        }

        // online softmax over this 16x32 slab (row spans 16 lanes in a half-wave)
#pragma unroll
        for (int i = 0; i < 8; ++i) {
            float tm = fmaxf(s0[i], s1[i]);
            for (int msk = 1; msk < 16; msk <<= 1) tm = fmaxf(tm, __shfl_xor(tm, msk, 32));
            float mnew = fmaxf(mrow[i], tm);
            float corr = __expf(mrow[i] - mnew);
            float p0 = __expf(s0[i] - mnew);
            float p1 = __expf(s1[i] - mnew);
            float rs = p0 + p1;
            for (int msk = 1; msk < 16; msk <<= 1) rs += __shfl_xor(rs, msk, 32);
            lrow[i] = lrow[i] * corr + rs;
            mrow[i] = mnew;
            o0[i] *= corr; o1[i] *= corr; o2[i] *= corr; o3[i] *= corr;
            int prow = i + hi * 8;
            Plds[wave * 512 + prow * 32 + ln]      = f2bf(p0);
            Plds[wave * 512 + prow * 32 + 16 + ln] = f2bf(p1);
        }
        asm volatile("s_wait_dscnt 0" ::: "memory");   // intra-wave LDS RAW on Plds

        Frag pa;
#pragma unroll
        for (int j = 0; j < 8; ++j) {
            int ka = 2 * j + ((j >= 4) ? 8 : 0) + (hi ? 8 : 0);
            pa.u[j] = *(const unsigned int*)&Plds[wave * 512 + ln * 32 + ka];
        }
#pragma unroll
        for (int nt = 0; nt < 4; ++nt) {
            Frag vb;
#pragma unroll
            for (int j = 0; j < 8; ++j) {
                int kk = 2 * j + ((j >= 4) ? 8 : 0) + (hi ? 8 : 0);
                unsigned int lo  = Vlds[buf][kk * 64 + nt * 16 + ln];
                unsigned int hi2 = Vlds[buf][(kk + 1) * 64 + nt * 16 + ln];
                vb.u[j] = lo | (hi2 << 16);
            }
            v8f* op = (nt == 0) ? &o0 : (nt == 1) ? &o1 : (nt == 2) ? &o2 : &o3;
            *op = __builtin_amdgcn_wmma_f32_16x16x32_bf16(false, pa.v, false, vb.v,
                                                          (short)0, *op, false, false);
        }
        buf ^= 1;
    }

#pragma unroll
    for (int i = 0; i < 8; ++i) {
        int row = q0 + i + hi * 8;
        float inv = 1.0f / lrow[i];
        size_t base = ((size_t)(b * PSEQ + row)) * DIMC + h * HDIM;
        attn_out[base + 0 * 16 + ln] = f2bf(o0[i] * inv);
        attn_out[base + 1 * 16 + ln] = f2bf(o1[i] * inv);
        attn_out[base + 2 * 16 + ln] = f2bf(o2[i] * inv);
        attn_out[base + 3 * 16 + ln] = f2bf(o3[i] * inv);
    }
}

// ---------------------------------------------------------------- launcher
extern "C" void kernel_launch(void* const* d_in, const int* in_sizes, int n_in,
                              void* d_out, int out_size, void* d_ws, size_t ws_size,
                              hipStream_t stream) {
    const float* x  = (const float*)d_in[0];
    const float* Wq = (const float*)d_in[1];
    const float* Wk = (const float*)d_in[2];
    const float* Wv = (const float*)d_in[3];
    const float* Wp = (const float*)d_in[4];
    const float* bp = (const float*)d_in[5];
    float* out = (float*)d_out;

    char* ws = (char*)d_ws;
    unsigned short* x_bf  = (unsigned short*)(ws + 0);                   // 16 MB
    unsigned short* w_qkv = (unsigned short*)(ws + (size_t)(16u << 20)); // 4 MB
    unsigned short* w_p   = (unsigned short*)(ws + (size_t)(20u << 20)); // 2 MB
    unsigned short* qkv   = (unsigned short*)(ws + (size_t)(22u << 20)); // 32 MB
    unsigned short* attn  = (unsigned short*)(ws + (size_t)(54u << 20)); // 16 MB
    float* kpart          = (float*)(ws + (size_t)(70u << 20));
    int*   kvidx          = (int*)(ws + (size_t)(70u << 20) + 1024);

    kdgqa_cvt_bf16<<<1024, 256, 0, stream>>>(x,  x_bf, MROWS * DIMC);
    kdgqa_cvt_bf16<<<512, 256, 0, stream>>>(Wq, w_qkv, DIMC * DIMC);
    kdgqa_cvt_bf16<<<256, 256, 0, stream>>>(Wk, w_qkv + DIMC * DIMC, 512 * DIMC);
    kdgqa_cvt_bf16<<<256, 256, 0, stream>>>(Wv, w_qkv + DIMC * DIMC + 512 * DIMC, 512 * DIMC);
    kdgqa_cvt_bf16<<<512, 256, 0, stream>>>(Wp, w_p, DIMC * DIMC);

    // qkv = x @ [Wq;Wk;Wv]^T  (bf16 out; q cols scaled by 1/sqrt(64))
    kdgqa_gemm_wmma<<<dim3(QKVN / 128, MROWS / 128), 256, 0, stream>>>(
        x_bf, w_qkv, qkv, nullptr, nullptr, MROWS, QKVN, DIMC, DIMC, 0.125f);

    kdgqa_knorm<<<64, 256, 0, stream>>>(qkv, kpart);
    kdgqa_alloc<<<1, 1, 0, stream>>>(kpart, kvidx);

    kdgqa_attn_wmma<<<1024, 256, 0, stream>>>(qkv, kvidx, attn);

    // out = attn @ Wp^T + bp  (f32 out)
    kdgqa_gemm_wmma<<<dim3(DIMC / 128, MROWS / 128), 256, 0, stream>>>(
        attn, w_p, nullptr, out, bp, MROWS, DIMC, DIMC, 0, 1.0f);
}
